// BeitSelfAttention_42417097015935
// MI455X (gfx1250) — matrix operand order repaired and checked by reading, not compiled, not run on hardware
//
#include <hip/hip_runtime.h>
#include <hip/hip_bf16.h>

// ---------------- types ----------------
typedef _Float16 f16x8  __attribute__((ext_vector_type(8)));
typedef _Float16 f16x16 __attribute__((ext_vector_type(16)));
typedef float    f32x8  __attribute__((ext_vector_type(8)));

// ---------------- problem constants ----------------
#define NB    8
#define SEQ   1569
#define DIM   768
#define NH    12
#define DH    64
#define BS    49
#define MB    32          // number of key/query blocks
#define NKV   6           // 3 local + 3 random
#define NKEY  295         // 1 cls + NKV*BS
#define NKP   320         // padded key count (mult of 32)
#define ROWS  (NB*SEQ)    // 12552
#define RTILE 785         // ceil(12552/16)
#define SCALE 0.125f      // 1/sqrt(64)

// workspace layout (bytes)
#define XH_OFF  0ull                      // f16 hidden  [12552*768]
#define WT_OFF  19279872ull               // f16 W^T     [3][768*768]  (N-major)
#define QH_OFF  22818816ull               // f16 q [B,NH,S,DH]
#define KH_OFF  42098688ull               // f16 k
#define VH_OFF  61378560ull               // f16 v
// total = 80,658,432 bytes

static __device__ __forceinline__ f32x8 wmma_f16(f16x16 a, f16x16 b, f32x8 c) {
  return __builtin_amdgcn_wmma_f32_16x16x32_f16(false, a, false, b, (short)0, c,
                                                false, false);
}
static __device__ __forceinline__ f16x16 join8(f16x8 lo, f16x8 hi) {
  return __builtin_shufflevector(lo, hi, 0,1,2,3,4,5,6,7,8,9,10,11,12,13,14,15);
}

// CDNA5 async copy: 16B/lane global -> LDS, tracked by ASYNCcnt.
// ISA: GLOBAL_LOAD_ASYNC_TO_LDS_B128  (LDS address in the first VGPR operand)
static __device__ __forceinline__ void async_ld_b128(unsigned lds_off, const void* g) {
  asm volatile("global_load_async_to_lds_b128 %0, %1, off"
               :: "v"(lds_off), "v"((unsigned long long)(uintptr_t)g)
               : "memory");
}
static __device__ __forceinline__ void wait_async0() {
  asm volatile("s_wait_asynccnt 0" ::: "memory");
}

// ---------------- kernel 1a: fp32 -> f16 hidden ----------------
__global__ void pack_x(const float* __restrict__ X, _Float16* __restrict__ Xh, int n) {
  for (int i = blockIdx.x * blockDim.x + threadIdx.x; i < n; i += gridDim.x * blockDim.x)
    Xh[i] = (_Float16)X[i];
}

// ---------------- kernel 1b: W (K,N) -> f16 W^T (N,K), 3 matrices ----------------
__global__ void pack_w(const float* __restrict__ Wq, const float* __restrict__ Wk,
                       const float* __restrict__ Wv, _Float16* __restrict__ Wt) {
  int i = blockIdx.x * blockDim.x + threadIdx.x;
  if (i >= 3 * DIM * DIM) return;
  int z = i / (DIM * DIM), r = i % (DIM * DIM);
  int n = r / DIM, k = r % DIM;
  const float* W = (z == 0) ? Wq : (z == 1) ? Wk : Wv;
  Wt[i] = (_Float16)W[k * DIM + n];
}

// ---------------- kernel 2: QKV projection GEMM (WMMA) ----------------
// grid (24 col-strips, 99 row-tile groups, 3 matrices), 256 thr = 8 waves.
// Each wave computes a 16x32 strip (two 16x16 tiles sharing the A fragment),
// fp32 accumulate, bias fused, f16 store in [B,NH,S,DH] layout.
__global__ __launch_bounds__(256) void qkv_gemm(
    const _Float16* __restrict__ Xh, const _Float16* __restrict__ Wt,
    const float* __restrict__ bq, const float* __restrict__ bv,
    _Float16* __restrict__ qh, _Float16* __restrict__ kh, _Float16* __restrict__ vh) {
  const int lane = threadIdx.x & 31;
  const int wave = threadIdx.x >> 5;
  const int rowTile = blockIdx.y * 8 + wave;
  if (rowTile >= RTILE) return;                 // uniform per wave
  const int z     = blockIdx.z;
  const int cBase = blockIdx.x * 32;
  const int rBase = rowTile * 16;
  const int l15   = lane & 15;
  const int hi    = lane >> 4;                  // 0 for lanes 0-15, 1 for 16-31

  const int arow = (rBase + l15 < ROWS) ? (rBase + l15) : (ROWS - 1);
  const _Float16* ap  = Xh + (size_t)arow * DIM + hi * 16;
  const _Float16* wz  = Wt + (size_t)z * DIM * DIM;
  const _Float16* bp0 = wz + (size_t)(cBase + l15) * DIM + hi * 16;
  const _Float16* bp1 = wz + (size_t)(cBase + 16 + l15) * DIM + hi * 16;

  f32x8 acc0 = {}, acc1 = {};
#pragma unroll 4
  for (int kk = 0; kk < DIM; kk += 32) {
    f16x8  a0 = *(const f16x8 *)(ap + kk);
    f16x8  a1 = *(const f16x8 *)(ap + kk + 8);
    f16x16 a  = join8(a0, a1);
    f16x16 b0 = *(const f16x16*)(bp0 + kk);
    f16x16 b1 = *(const f16x16*)(bp1 + kk);
    acc0 = wmma_f16(a, b0, acc0);
    acc1 = wmma_f16(a, b1, acc1);
  }

  _Float16* dst = (z == 0) ? qh : (z == 1) ? kh : vh;
#pragma unroll
  for (int t = 0; t < 2; t++) {
    const int n = cBase + t * 16 + l15;
    float bias = (z == 0) ? bq[n] : (z == 2) ? bv[n] : 0.0f;
    const int h = n >> 6, d = n & 63;
    const f32x8 acc = t ? acc1 : acc0;
#pragma unroll
    for (int j = 0; j < 8; j++) {
      int r = rBase + j + hi * 8;
      if (r < ROWS) {
        int b_ = r / SEQ, s = r % SEQ;
        dst[(((size_t)b_ * NH + h) * SEQ + s) * DH + d] = (_Float16)(acc[j] + bias);
      }
    }
  }
}

// ---------------- kernel 3: CLS-query attention (1 row vs all S keys) ----------------
__global__ __launch_bounds__(256) void cls_attn(
    const _Float16* __restrict__ qh, const _Float16* __restrict__ kh,
    const _Float16* __restrict__ vh, const float* __restrict__ rel_table,
    const int* __restrict__ rpi, float* __restrict__ out) {
  __shared__ float qf[DH];
  __shared__ float simS[SEQ];
  __shared__ float red[256];
  const int bh = blockIdx.x, b = bh / NH, h = bh % NH;
  const size_t ho = (size_t)bh * SEQ * DH;
  const int tid = threadIdx.x;

  if (tid < DH) qf[tid] = (float)qh[ho + tid];
  __syncthreads();

  float lmax = -1e30f;
  for (int j = tid; j < SEQ; j += 256) {
    const _Float16* kr = kh + ho + (size_t)j * DH;
    float dot = 0.f;
#pragma unroll
    for (int d = 0; d < DH; d++) dot += qf[d] * (float)kr[d];
    float v = dot * SCALE + rel_table[rpi[j] * NH + h];
    simS[j] = v;
    lmax = fmaxf(lmax, v);
  }
  red[tid] = lmax; __syncthreads();
  for (int s = 128; s > 0; s >>= 1) { if (tid < s) red[tid] = fmaxf(red[tid], red[tid + s]); __syncthreads(); }
  float mx = red[0]; __syncthreads();

  float ls = 0.f;
  for (int j = tid; j < SEQ; j += 256) { float e = __expf(simS[j] - mx); simS[j] = e; ls += e; }
  red[tid] = ls; __syncthreads();
  for (int s = 128; s > 0; s >>= 1) { if (tid < s) red[tid] += red[tid + s]; __syncthreads(); }
  float inv = 1.f / red[0];

  if (tid < DH) {
    float o = 0.f;
    for (int j = 0; j < SEQ; j++) o += simS[j] * (float)vh[ho + (size_t)j * DH + tid];
    out[((size_t)b * SEQ) * DIM + h * DH + tid] = o * inv;
  }
}

// ---------------- kernel 4: block-sparse attention (WMMA + async LDS gather) ----------------
// grid (M=32, NH=12, B=8), 256 thr = 8 waves, dynamic LDS = 152832 B:
//   Qs  [64][64]  f16 @0       (8192 B)
//   Kg  [320][64] f16 @8192    (40960 B)   (later aliased by Pf [64][320] f16)
//   VgT [64][320] f16 @49152   (40960 B)
//   sim [49][320] f32 @90112   (62720 B)
__global__ __launch_bounds__(256) void block_attn(
    const _Float16* __restrict__ qh, const _Float16* __restrict__ kh,
    const _Float16* __restrict__ vh, const float* __restrict__ rel_table,
    const int* __restrict__ rpi, const int* __restrict__ rand_idx,
    float* __restrict__ out) {
  extern __shared__ char smem[];
  _Float16* Qs  = (_Float16*)(smem);
  _Float16* Kg  = (_Float16*)(smem + 8192);
  _Float16* VgT = (_Float16*)(smem + 49152);
  float*    sim = (float*)   (smem + 90112);
  __shared__ int sidx[NKV];

  const int m = blockIdx.x, h = blockIdx.y, b = blockIdx.z;
  const int tid = threadIdx.x;

  if (tid < NKV) {
    int v;
    if      (tid == 0) v = (m + MB - 1) & (MB - 1);
    else if (tid == 1) v = m;
    else if (tid == 2) v = (m + 1) & (MB - 1);
    else               v = rand_idx[m * 3 + (tid - 3)];
    sidx[tid] = v;
  }
  __syncthreads();

  const size_t ho = (size_t)(b * NH + h) * SEQ * DH;
  const _Float16* qh_h = qh + ho;
  const _Float16* kh_h = kh + ho;
  const _Float16* vh_h = vh + ho;

  // LDS byte offsets (low 32 bits of the shared-aperture generic address)
  const unsigned qs_lds = (unsigned)(uintptr_t)(void*)Qs;
  const unsigned kg_lds = (unsigned)(uintptr_t)(void*)Kg;

  // zero padding rows with plain DS stores: Q rows 49..63, K rows 295..319
  for (int i = tid; i < (64 - BS) * (DH / 2); i += 256)      // 480 dwords
    ((unsigned*)(smem + BS * 128))[i] = 0;
  for (int i = tid; i < (NKP - NKEY) * (DH / 2); i += 256)   // 800 dwords
    ((unsigned*)(smem + 8192 + NKEY * 128))[i] = 0;

  // async gather Q rows 0..48 (8 x 16B chunks per row), global -> LDS direct
  for (int i = tid; i < BS * 8; i += 256) {
    int row = i >> 3, seg = i & 7;
    async_ld_b128(qs_lds + row * 128 + seg * 16,
                  qh_h + (size_t)(1 + m * BS + row) * DH + seg * 8);
  }
  // async gather K rows 0..294: row 0 = cls, rows 1.. = gathered blocks
  for (int i = tid; i < NKEY * 8; i += 256) {
    int row = i >> 3, seg = i & 7;
    int t = (row == 0) ? 0 : 1 + sidx[(row - 1) / BS] * BS + ((row - 1) % BS);
    async_ld_b128(kg_lds + row * 128 + seg * 16,
                  kh_h + (size_t)t * DH + seg * 8);
  }
  // gather V transposed (VALU path; transpose can't use async B128): VgT[d][key]
  for (int i = tid; i < NKP * DH; i += 256) {
    int key = i >> 6, d = i & 63;
    _Float16 v = (_Float16)0.0f;
    if (key < NKEY) {
      int t = (key == 0) ? 0 : 1 + sidx[(key - 1) / BS] * BS + ((key - 1) % BS);
      v = vh_h[(size_t)t * DH + d];
    }
    VgT[d * NKP + key] = v;
  }
  wait_async0();
  __syncthreads();

  const int lane = tid & 31, wave = tid >> 5;
  const int l15 = lane & 15, hi = lane >> 4;

  // S = Q * Kg^T  (64x320, fp32 accumulate)  : 4 Mtiles x 20 Ntiles, K=64
  for (int job = wave; job < 4 * 20; job += 8) {
    int mt = job / 20, nt = job % 20;
    const _Float16* ap = Qs + (mt * 16 + l15) * DH + hi * 16;
    const _Float16* bp = Kg + (nt * 16 + l15) * DH + hi * 16;
    f32x8 acc = {};
#pragma unroll
    for (int ks = 0; ks < 2; ks++) {
      f16x8  a0 = *(const f16x8 *)(ap + ks * 32);
      f16x8  a1 = *(const f16x8 *)(ap + ks * 32 + 8);
      f16x16 bb = *(const f16x16*)(bp + ks * 32);
      acc = wmma_f16(join8(a0, a1), bb, acc);
    }
#pragma unroll
    for (int j = 0; j < 8; j++) {
      int r = mt * 16 + j + hi * 8;
      if (r < BS) sim[r * NKP + nt * 16 + l15] = acc[j];
    }
  }
  __syncthreads();

  // P (f16, softmax result) aliases Kg; zero it first (padding must be 0)
  _Float16* Pf = Kg;
  for (int i = tid; i < 64 * NKP / 2; i += 256) ((unsigned*)Pf)[i] = 0;
  __syncthreads();

  // fused scale + rel-pos bias + softmax, one wave per query row
  for (int r = wave; r < BS; r += 8) {
    const int  tq = 1 + m * BS + r;
    const int* rpirow = rpi + (size_t)tq * SEQ;
    float vals[10];
    float mx = -1e30f;
    int cnt = 0;
    for (int c = lane; c < NKEY; c += 32) {
      int tk = (c == 0) ? 0 : 1 + sidx[(c - 1) / BS] * BS + ((c - 1) % BS);
      float v = sim[r * NKP + c] * SCALE + rel_table[rpirow[tk] * NH + h];
      vals[cnt++] = v;
      mx = fmaxf(mx, v);
    }
    for (int o = 16; o > 0; o >>= 1) mx = fmaxf(mx, __shfl_xor(mx, o, 32));
    float sum = 0.f; cnt = 0;
    for (int c = lane; c < NKEY; c += 32) {
      float e = __expf(vals[cnt] - mx); vals[cnt++] = e; sum += e;
    }
    for (int o = 16; o > 0; o >>= 1) sum += __shfl_xor(sum, o, 32);
    float inv = 1.f / sum;
    cnt = 0;
    for (int c = lane; c < NKEY; c += 32) Pf[r * NKP + c] = (_Float16)(vals[cnt++] * inv);
  }
  __syncthreads();

  // O = P * Vg  (64x64, K=320) : 4 Mtiles x 4 Ntiles, 10 k-steps
  for (int job = wave; job < 16; job += 8) {
    int mt = job >> 2, nt = job & 3;
    const _Float16* ap = Pf  + (mt * 16 + l15) * NKP + hi * 16;
    const _Float16* bp = VgT + (nt * 16 + l15) * NKP + hi * 16;
    f32x8 acc = {};
#pragma unroll
    for (int ks = 0; ks < NKP / 32; ks++) {
      f16x8  a0 = *(const f16x8 *)(ap + ks * 32);
      f16x8  a1 = *(const f16x8 *)(ap + ks * 32 + 8);
      f16x16 bb = *(const f16x16*)(bp + ks * 32);
      acc = wmma_f16(join8(a0, a1), bb, acc);
    }
#pragma unroll
    for (int j = 0; j < 8; j++) {
      int r = mt * 16 + j + hi * 8;
      if (r < BS) {
        int tok = 1 + m * BS + r;
        out[((size_t)b * SEQ + tok) * DIM + h * DH + nt * 16 + l15] = acc[j];
      }
    }
  }
}

// ---------------- launcher ----------------
extern "C" void kernel_launch(void* const* d_in, const int* in_sizes, int n_in,
                              void* d_out, int out_size, void* d_ws, size_t ws_size,
                              hipStream_t stream) {
  const float* hidden    = (const float*)d_in[0];
  const float* Wq        = (const float*)d_in[1];
  const float* bq        = (const float*)d_in[2];
  const float* Wk        = (const float*)d_in[3];
  const float* Wv        = (const float*)d_in[4];
  const float* bv        = (const float*)d_in[5];
  const float* rel_table = (const float*)d_in[6];
  const int*   rpi       = (const int*)d_in[7];
  const int*   rand_idx  = (const int*)d_in[8];
  float* out = (float*)d_out;

  char* ws = (char*)d_ws;                 // needs ~81 MB of scratch
  _Float16* Xh = (_Float16*)(ws + XH_OFF);
  _Float16* Wt = (_Float16*)(ws + WT_OFF);
  _Float16* qh = (_Float16*)(ws + QH_OFF);
  _Float16* kh = (_Float16*)(ws + KH_OFF);
  _Float16* vh = (_Float16*)(ws + VH_OFF);

  const int nX = ROWS * DIM;
  pack_x<<<(nX + 255) / 256, 256, 0, stream>>>(hidden, Xh, nX);
  pack_w<<<(3 * DIM * DIM + 255) / 256, 256, 0, stream>>>(Wq, Wk, Wv, Wt);

  qkv_gemm<<<dim3(DIM / 32, (RTILE + 7) / 8, 3), 256, 0, stream>>>(
      Xh, Wt, bq, bv, qh, kh, vh);

  cls_attn<<<NB * NH, 256, 0, stream>>>(qh, kh, vh, rel_table, rpi, out);

  block_attn<<<dim3(MB, NH, NB), 256, 152832, stream>>>(
      qh, kh, vh, rel_table, rpi, rand_idx, out);
}
